// _WindowAttention_17678085391002
// MI455X (gfx1250) — matrix operand order repaired and checked by reading, not compiled, not run on hardware
//
#include <hip/hip_runtime.h>
#include <stdint.h>

// ---------------------------------------------------------------------------
// Swin window attention for MI455X (gfx1250), bf16 WMMA path.
//   tokens N=64, DIM=128, H=4 heads, HD=32, 8192 windows, nW=64 masks.
//   One workgroup (8 wave32) per window; ~134.6 KB LDS -> 2 WGs per WGP.
// ---------------------------------------------------------------------------

typedef __attribute__((ext_vector_type(16))) __bf16 bh16;
typedef __attribute__((ext_vector_type(8)))  float  v8f;

#define TOK   64
#define XDIM  128
#define NH    4
#define HD    32
#define SCALE 0.17677669529663687f   // 32^-0.5

// ---- LDS layout (bytes). Odd uint strides kill bank conflicts. ----
#define QKV_STRIDE_U 197                       // uints per qkv row (384 bf16 + pad)
#define LDS_QKV   0
#define SZ_QKV    (64 * QKV_STRIDE_U * 4)      // 50432
#define S_STRIDE  65                            // floats per S row
#define LDS_S     (LDS_QKV + SZ_QKV)
#define SZ_S      (4 * 64 * S_STRIDE * 4)      // 66560 (P bf16 overlays S rows)
#define XO_STRIDE_U 65                          // x (bf16) reused as O (bf16)
#define LDS_XO    (LDS_S + SZ_S)
#define SZ_XO     (64 * XO_STRIDE_U * 4)       // 16640
#define LDS_RS    (LDS_XO + SZ_XO)
#define SZ_RS     (4 * 64 * 4)                 // 1024 (1/rowsum per head/row)
#define SMEM_TOTAL (LDS_RS + SZ_RS)            // 134656 B -> 2 workgroups/WGP

__device__ __forceinline__ unsigned short f2bf(float f) {
  unsigned u = __float_as_uint(f);
  u += 0x7FFFu + ((u >> 16) & 1u);             // round-to-nearest-even
  return (unsigned short)(u >> 16);
}
__device__ __forceinline__ unsigned pack2(float lo, float hi) {
  return (unsigned)f2bf(lo) | ((unsigned)f2bf(hi) << 16);
}

union Frag { unsigned u[8]; bh16 v; };

__device__ __forceinline__ v8f wmma_bf16(const Frag& a, const Frag& b, v8f c) {
  // (neg_a, A, neg_b, B, c_mod, C, reuse_a, reuse_b)
  return __builtin_amdgcn_wmma_f32_16x16x32_bf16(false, a.v, false, b.v,
                                                 (short)0, c, false, false);
}

// ---------------------------------------------------------------------------
// Prep: f32 weights -> bf16 in d_ws; build rel-pos bias table bias[h][n][m].
// ---------------------------------------------------------------------------
__global__ void _wa_prep(const float* __restrict__ qkv_w,
                         const float* __restrict__ proj_w,
                         const float* __restrict__ rel_pos_bias,
                         unsigned short* __restrict__ wqkv,
                         unsigned short* __restrict__ wproj,
                         float* __restrict__ biasT) {
  int tid = blockIdx.x * 256 + threadIdx.x;
  if (tid < 49152) {                       // qkv_w [384][128]
    wqkv[tid] = f2bf(qkv_w[tid]);
  } else if (tid < 65536) {                // proj_w [128][128]
    int t = tid - 49152;
    wproj[t] = f2bf(proj_w[t]);
  } else if (tid < 81920) {                // bias[h][n][m]
    int t = tid - 65536;
    int h = t >> 12, nm = t & 4095, n = nm >> 6, m = nm & 63;
    int di = (n >> 3) - (m >> 3) + 7;
    int dj = (n & 7)  - (m & 7)  + 7;
    biasT[t] = rel_pos_bias[(di * 15 + dj) * NH + h];
  }
}

// ---------------------------------------------------------------------------
// Main: one workgroup (8 wave32) per window.
// ---------------------------------------------------------------------------
__global__ __launch_bounds__(256)
void _wa_main(const float* __restrict__ x, const float* __restrict__ mask,
              const float* __restrict__ qkv_b, const float* __restrict__ proj_b,
              const unsigned* __restrict__ WqkvU,   // bf16 [384][128] as uints
              const unsigned* __restrict__ WprojU,  // bf16 [128][128] as uints
              const float* __restrict__ biasT,      // [4][64][64]
              float* __restrict__ out) {
  extern __shared__ char smem[];
  unsigned*       qkv_u = (unsigned*)(smem + LDS_QKV);
  unsigned short* qkv_h = (unsigned short*)(smem + LDS_QKV);
  float*          S     = (float*)(smem + LDS_S);
  unsigned*       P_u   = (unsigned*)(smem + LDS_S);   // bf16 P overlays S rows
  unsigned*       xo_u  = (unsigned*)(smem + LDS_XO);
  unsigned short* xo_h  = (unsigned short*)(smem + LDS_XO);
  float*          rs    = (float*)(smem + LDS_RS);

  const int b    = blockIdx.x;
  const int tid  = threadIdx.x;
  const int wave = tid >> 5, lane = tid & 31;
  const int half = lane >> 4, hr = lane & 15;

  const float* xw    = x    + (size_t)b * (TOK * XDIM);
  const float* maskw = mask + (size_t)(b & 63) * (TOK * TOK);
  __builtin_prefetch(maskw, 0, 0);   // global_prefetch_b8: warm L2 for mask

  // ---- phase 1: x[64][128] f32 -> LDS bf16 (padded rows) ----
  const float4* x4 = (const float4*)xw;
  for (int i = tid; i < TOK * XDIM / 4; i += 256) {
    float4 vv = x4[i];
    int f = i * 4, row = f >> 7, col = f & 127;
    xo_u[row * XO_STRIDE_U + (col >> 1)]     = pack2(vv.x, vv.y);
    xo_u[row * XO_STRIDE_U + (col >> 1) + 1] = pack2(vv.z, vv.w);
  }
  __syncthreads();

  // ---- phase 2: qkv = x @ qkv_w.T + qkv_b  (M=64, N=384, K=128) ----
  {
    const int mt = wave & 3;            // M tile
    const int n0 = (wave >> 2) * 12;    // 12 N tiles per wave
    const int arow = mt * 16 + hr;
    Frag A[4];
#pragma unroll
    for (int kt = 0; kt < 4; ++kt)
#pragma unroll
      for (int i = 0; i < 8; ++i) {
        int kb = kt * 32 + (half << 4) + (i << 1);
        A[kt].u[i] = xo_u[arow * XO_STRIDE_U + (kb >> 1)];
      }
    for (int nt = n0; nt < n0 + 12; ++nt) {
      int col = nt * 16 + hr;
      float bias = qkv_b[col];
      v8f acc = {bias, bias, bias, bias, bias, bias, bias, bias};
#pragma unroll
      for (int kt = 0; kt < 4; ++kt) {
        const uint4* wp = (const uint4*)(WqkvU + col * 64 + kt * 16 + half * 8);
        uint4 b0 = wp[0], b1 = wp[1];
        Frag B;
        B.u[0] = b0.x; B.u[1] = b0.y; B.u[2] = b0.z; B.u[3] = b0.w;
        B.u[4] = b1.x; B.u[5] = b1.y; B.u[6] = b1.z; B.u[7] = b1.w;
        acc = wmma_bf16(A[kt], B, acc);
      }
#pragma unroll
      for (int r = 0; r < 8; ++r) {
        int row = mt * 16 + half * 8 + r;   // D layout: M = r + 8*half
        qkv_h[row * (QKV_STRIDE_U * 2) + col] = f2bf(acc[r]);
      }
    }
  }
  __syncthreads();

  // ---- phase 3: S = q_h @ k_h^T  (per head, K=32 -> single WMMA/tile) ----
  {
    const int h = wave & 3, p = wave >> 2;
    float* Sh = S + h * 64 * S_STRIDE;
    for (int mti = 0; mti < 2; ++mti) {
      int mt = p * 2 + mti;
      int arow = mt * 16 + hr;
      Frag Aq;
#pragma unroll
      for (int i = 0; i < 8; ++i) {
        int kb = h * 32 + (half << 4) + (i << 1);
        Aq.u[i] = qkv_u[arow * QKV_STRIDE_U + (kb >> 1)];
      }
#pragma unroll
      for (int nt = 0; nt < 4; ++nt) {
        int colTok = nt * 16 + hr;
        const unsigned* kp = qkv_u + colTok * QKV_STRIDE_U + 64 + h * 16 + half * 8;
        Frag Bk;
#pragma unroll
        for (int i = 0; i < 8; ++i) Bk.u[i] = kp[i];
        v8f acc = {0, 0, 0, 0, 0, 0, 0, 0};
        acc = wmma_bf16(Aq, Bk, acc);
#pragma unroll
        for (int r = 0; r < 8; ++r)
          Sh[(mt * 16 + half * 8 + r) * S_STRIDE + colTok] = acc[r];
      }
    }
  }
  __syncthreads();

  // ---- phase 4: softmax rows. Pass 1 fuses scale+bias+mask in place;
  //      pass 2 exp + bf16-pack, P overlaying the same S row (safe: bf16
  //      P[m] at byte 2m never touches unread f32 S[m'>=m] at byte 4m'). ----
  {
    const int h = wave & 3, p = wave >> 2;
    int row = p * 32 + lane;                         // each lane owns one row
    float* srow = S + (h * 64 + row) * S_STRIDE;
    const float4* brow = (const float4*)(biasT + (h * 64 + row) * 64);
    const float4* mrow = (const float4*)(maskw + row * 64);
    float mx = -3.4e38f;
    for (int m4 = 0; m4 < 16; ++m4) {
      float4 bv = brow[m4], mv = mrow[m4];
      float v0 = srow[m4 * 4 + 0] * SCALE + bv.x + mv.x;
      float v1 = srow[m4 * 4 + 1] * SCALE + bv.y + mv.y;
      float v2 = srow[m4 * 4 + 2] * SCALE + bv.z + mv.z;
      float v3 = srow[m4 * 4 + 3] * SCALE + bv.w + mv.w;
      srow[m4 * 4 + 0] = v0; srow[m4 * 4 + 1] = v1;
      srow[m4 * 4 + 2] = v2; srow[m4 * 4 + 3] = v3;
      mx = fmaxf(mx, fmaxf(fmaxf(v0, v1), fmaxf(v2, v3)));
    }
    float sum = 0.f;
    unsigned short* prow = (unsigned short*)srow;    // P overlay
    for (int m = 0; m < 64; ++m) {
      float e = __expf(srow[m] - mx);
      unsigned short eb = f2bf(e);
      prow[m] = eb;                                  // unnormalized bf16 P
      sum += __uint_as_float(((unsigned)eb) << 16);  // sum what WMMA will sum
    }
    rs[h * 64 + row] = 1.0f / sum;                   // fold into P@V epilogue
  }
  __syncthreads();

  // ---- phase 5: O_h = (P @ v_h) * (1/rowsum)  (M=64, N=32, K=64) ----
  {
    const int h = wave & 3, p = wave >> 2;
    for (int mti = 0; mti < 2; ++mti) {
      int mt = p * 2 + mti;
      int arow = mt * 16 + hr;
      Frag Ap[2];
#pragma unroll
      for (int kt = 0; kt < 2; ++kt)
#pragma unroll
        for (int i = 0; i < 8; ++i) {
          int kb = kt * 32 + (half << 4) + (i << 1);
          Ap[kt].u[i] = P_u[(h * 64 + arow) * S_STRIDE + (kb >> 1)];
        }
#pragma unroll
      for (int nt = 0; nt < 2; ++nt) {
        int coln = nt * 16 + hr;                     // head-dim column
        v8f acc = {0, 0, 0, 0, 0, 0, 0, 0};
#pragma unroll
        for (int kt = 0; kt < 2; ++kt) {
          Frag Bv;
          int kbase = kt * 32 + half * 16;
#pragma unroll
          for (int i = 0; i < 8; ++i) {              // B pair = (K=2i, 2i+1)
            int t0 = kbase + 2 * i;
            unsigned lo = qkv_h[t0 * (QKV_STRIDE_U * 2) + 256 + h * 32 + coln];
            unsigned hi = qkv_h[(t0 + 1) * (QKV_STRIDE_U * 2) + 256 + h * 32 + coln];
            Bv.u[i] = lo | (hi << 16);
          }
          acc = wmma_bf16(Ap[kt], Bv, acc);
        }
#pragma unroll
        for (int r = 0; r < 8; ++r) {
          int row = mt * 16 + half * 8 + r;
          float val = acc[r] * rs[h * 64 + row];
          xo_h[row * (XO_STRIDE_U * 2) + h * 32 + coln] = f2bf(val); // O over x
        }
      }
    }
  }
  __syncthreads();

  // ---- phase 6: out = O @ proj_w.T + proj_b  (M=64, N=128, K=128) ----
  {
    const int mt = wave & 3;
    const int n0 = (wave >> 2) * 4;
    const int arow = mt * 16 + hr;
    Frag A[4];
#pragma unroll
    for (int kt = 0; kt < 4; ++kt)
#pragma unroll
      for (int i = 0; i < 8; ++i) {
        int kb = kt * 32 + (half << 4) + (i << 1);
        A[kt].u[i] = xo_u[arow * XO_STRIDE_U + (kb >> 1)];
      }
    float* outw = out + (size_t)b * (TOK * XDIM);
#pragma unroll
    for (int nt = n0; nt < n0 + 4; ++nt) {
      int col = nt * 16 + hr;
      float bias = proj_b[col];
      v8f acc = {bias, bias, bias, bias, bias, bias, bias, bias};
#pragma unroll
      for (int kt = 0; kt < 4; ++kt) {
        const uint4* wp = (const uint4*)(WprojU + col * 64 + kt * 16 + half * 8);
        uint4 b0 = wp[0], b1 = wp[1];
        Frag B;
        B.u[0] = b0.x; B.u[1] = b0.y; B.u[2] = b0.z; B.u[3] = b0.w;
        B.u[4] = b1.x; B.u[5] = b1.y; B.u[6] = b1.z; B.u[7] = b1.w;
        acc = wmma_bf16(A[kt], B, acc);
      }
#pragma unroll
      for (int r = 0; r < 8; ++r)
        outw[(mt * 16 + half * 8 + r) * XDIM + col] = acc[r];
    }
  }
}

// ---------------------------------------------------------------------------
extern "C" void kernel_launch(void* const* d_in, const int* in_sizes, int n_in,
                              void* d_out, int out_size, void* d_ws, size_t ws_size,
                              hipStream_t stream) {
  const float* x      = (const float*)d_in[0];
  const float* mask   = (const float*)d_in[1];
  const float* qkv_w  = (const float*)d_in[2];
  const float* qkv_b  = (const float*)d_in[3];
  const float* proj_w = (const float*)d_in[4];
  const float* proj_b = (const float*)d_in[5];
  const float* rel    = (const float*)d_in[6];

  char* ws = (char*)d_ws;
  unsigned short* wqkv  = (unsigned short*)ws;             // 98304 B
  unsigned short* wproj = (unsigned short*)(ws + 98304);   // 32768 B
  float*          biasT = (float*)(ws + 131072);           // 65536 B

  (void)hipFuncSetAttribute((const void*)_wa_main,
                            hipFuncAttributeMaxDynamicSharedMemorySize,
                            SMEM_TOTAL);

  _wa_prep<<<320, 256, 0, stream>>>(qkv_w, proj_w, rel, wqkv, wproj, biasT);

  int nwin = in_sizes[0] / (TOK * XDIM);                   // 8192 windows
  _wa_main<<<nwin, 256, SMEM_TOTAL, stream>>>(
      x, mask, qkv_b, proj_b,
      (const unsigned*)wqkv, (const unsigned*)wproj, biasT, (float*)d_out);
  (void)n_in; (void)out_size; (void)ws_size;
}
